// TwoStageAttentionLayer_21045339750737
// MI455X (gfx1250) — matrix operand order repaired and checked by reading, not compiled, not run on hardware
//
#include <hip/hip_runtime.h>
#include <hip/hip_bf16.h>
#include <math.h>

// TwoStageAttentionLayer for MI455X (gfx1250): all GEMMs via v_wmma_f32_16x16x32_bf16.
// Weights are pre-packed once per launch into bf16 WMMA B-fragment layout so the
// hot loops do only b128 loads. A-fragments load as float4/b128 runs.
// Shapes: B=16, TS_D=128, SEG=64, DM=256, H=8, E=32, FACTOR=10, DFF=1024.

typedef __attribute__((ext_vector_type(16))) __bf16 v16bf;
typedef __attribute__((ext_vector_type(8)))  __bf16 v8bf;
typedef __attribute__((ext_vector_type(8)))  float  v8f;
typedef __attribute__((ext_vector_type(4)))  float  f4;

#define DMODEL 256

__device__ __forceinline__ v8f wmma_bf16(v16bf a, v16bf b, v8f c) {
  return __builtin_amdgcn_wmma_f32_16x16x32_bf16(false, a, false, b, (short)0, c, false, false);
}

__device__ __forceinline__ v16bf cat8(v8bf lo, v8bf hi) {
  return __builtin_shufflevector(lo, hi, 0,1,2,3,4,5,6,7,8,9,10,11,12,13,14,15);
}

// A-fragment from LDS: two 8-elt runs at p and p+16 (both 16B aligned).
__device__ __forceinline__ v16bf lds_a_frag(const __bf16* p) {
  return cat8(*(const v8bf*)p, *(const v8bf*)(p + 16));
}
// B fragment from LDS: two 8-elt runs at p and p+16 (K^T rows, V^T rows).
__device__ __forceinline__ v16bf lds_b_contig(const __bf16* p) {
  return cat8(*(const v8bf*)p, *(const v8bf*)(p + 16));
}
// A-fragment from global f32: runs at p..p+7 and p+16..p+23, converted to bf16.
__device__ __forceinline__ v16bf gbl_a_frag_f32(const float* p, bool ok) {
  v16bf r;
  if (ok) {
    f4 x0 = *(const f4*)(p);
    f4 x1 = *(const f4*)(p + 4);
    f4 x2 = *(const f4*)(p + 16);
    f4 x3 = *(const f4*)(p + 20);
#pragma unroll
    for (int i = 0; i < 4; ++i) {
      r[i]      = (__bf16)x0[i];
      r[4 + i]  = (__bf16)x1[i];
      r[8 + i]  = (__bf16)x2[i];
      r[12 + i] = (__bf16)x3[i];
    }
  } else {
#pragma unroll
    for (int i = 0; i < 16; ++i) r[i] = (__bf16)0.0f;
  }
  return r;
}
// B-fragment from packed bf16 weights: 16 contiguous bf16 per lane.
__device__ __forceinline__ v16bf pk_b_frag(const __bf16* P, int kc, int NT, int nt, int lane) {
  const __bf16* p = P + ((kc * NT + nt) << 9) + (lane << 4);
  return cat8(*(const v8bf*)p, *(const v8bf*)(p + 8));
}

__device__ __forceinline__ float gelu_exact(float x) {
  return 0.5f * x * (1.0f + erff(x * 0.70710678118654752f));
}

// ---------------------------------------------------------------------------
// Pack f32 W (K x N, row-major) into bf16 WMMA B-fragment layout:
// P[((kc*NT + nt)*32 + slot)*16 + i] = W[kc*32 + (slot>>4)*8 + (i&7) + (i>>3)*16][nt*16 + (slot&15)]
// ---------------------------------------------------------------------------
__global__ __launch_bounds__(256) void pack_w_kernel(const float* __restrict__ W,
                                                     __bf16* __restrict__ P, int N) {
  int idx = blockIdx.x * 256 + threadIdx.x;       // total = K*N, grid covers exactly
  int i = idx & 15;
  int slot = (idx >> 4) & 31;
  int rest = idx >> 9;
  int NT = N >> 4;
  int nt = rest % NT;
  int kc = rest / NT;
  int k = kc * 32 + ((slot >> 4) << 3) + (i & 7) + ((i >> 3) << 4);
  int col = nt * 16 + (slot & 15);
  P[idx] = (__bf16)W[k * N + col];
}

// ---------------------------------------------------------------------------
// Fused per-(batch, head) attention. Padded K rows -> -inf scores; padded V
// rows multiplied by P==0; padded Q rows never stored.
// ---------------------------------------------------------------------------
template<int LPAD, int SPAD>
__global__ __launch_bounds__(256) void attn_kernel(
    const float* __restrict__ qin_base, int q_bmod, int q_bstride, int Lr,
    const float* __restrict__ kv_base, int kv_bstride, int Sr,
    const __bf16* __restrict__ Wqp, const float* __restrict__ bqv,
    const __bf16* __restrict__ Wkp, const float* __restrict__ bkv,
    const __bf16* __restrict__ Wvp, const float* __restrict__ bvv,
    float* __restrict__ outp, int ob, int oh, int ol, float scale)
{
  __shared__ __align__(16) __bf16 sQ[LPAD * 32];
  __shared__ __align__(16) __bf16 sK[SPAD * 32];
  __shared__ __align__(16) __bf16 sVt[32 * SPAD];   // transposed: [e][s]
  __shared__ __align__(16) float  sP[LPAD * SPAD];
  __shared__ __align__(16) __bf16 sPb[LPAD * SPAD];

  const int tid = threadIdx.x, lane = tid & 31, wave = tid >> 5;
  const int bq = blockIdx.x, h = blockIdx.y;
  const float* qin  = qin_base + (long)(bq % q_bmod) * q_bstride;
  const float* kvin = kv_base  + (long)bq * kv_bstride;

  const int mrow  = lane & 15;
  const int khalf = (lane >> 4) << 3;  // A-frag k offset (0 or 8)
  const int kb16  = (lane >> 4) << 3;  // B-frag k offset (0 or 8)
  const int mhalf = (lane >> 4) << 3;  // C/D row offset (0 or 8)

  // ---- phase 1: per-head Q/K/V projections (K=256 -> 8 wmma steps) ----
  constexpr int QT = (LPAD / 16) * 2, KT = (SPAD / 16) * 2;
  for (int t = wave; t < QT + 2 * KT; t += 8) {
    const float* src; const __bf16* Wp; const float* bias; int rows; int mt; int which;
    if (t < QT)           { mt = t;           src = qin;  Wp = Wqp; bias = bqv; rows = Lr; which = 0; }
    else if (t < QT + KT) { mt = t - QT;      src = kvin; Wp = Wkp; bias = bkv; rows = Sr; which = 1; }
    else                  { mt = t - QT - KT; src = kvin; Wp = Wvp; bias = bvv; rows = Sr; which = 2; }
    const int m0 = (mt >> 1) * 16, n0 = (mt & 1) * 16;
    const int m = m0 + mrow;
    const bool mok = m < rows;
    const int n = n0 + mrow;
    const int nt = h * 2 + (n0 >> 4);  // global 16-col tile in 256-wide W
    v8f acc = {};
    for (int kc = 0; kc < 8; ++kc) {
      v16bf a = gbl_a_frag_f32(src + m * DMODEL + kc * 32 + khalf, mok);
      v16bf b = pk_b_frag(Wp, kc, 16, nt, lane);
      acc = wmma_bf16(a, b, acc);
    }
    float bb = bias[h * 32 + n];
#pragma unroll
    for (int j = 0; j < 8; ++j) {
      int rr = m0 + j + mhalf;
      float v = acc[j] + bb;
      if (which == 0)      sQ[rr * 32 + n] = (__bf16)v;
      else if (which == 1) sK[rr * 32 + n] = (__bf16)v;
      else                 sVt[n * SPAD + rr] = (__bf16)v;
    }
  }
  __syncthreads();

  // ---- phase 2: scores = Q @ K^T * scale (K-dim = E = 32, one wmma) ----
  constexpr int LT = LPAD / 16, ST = SPAD / 16;
  for (int t = wave; t < LT * ST; t += 8) {
    const int m0 = (t / ST) * 16, s0 = (t % ST) * 16;
    const int m = m0 + mrow, s = s0 + mrow;
    v16bf a = lds_a_frag(sQ + m * 32 + khalf);
    v16bf b = lds_b_contig(sK + s * 32 + kb16);   // K^T: contiguous row of K
    v8f acc = {};
    acc = wmma_bf16(a, b, acc);
    const bool sok = s < Sr;
#pragma unroll
    for (int j = 0; j < 8; ++j)
      sP[(m0 + j + mhalf) * SPAD + s] = sok ? acc[j] * scale : -3.0e38f;
  }
  __syncthreads();

  // ---- phase 3: row softmax (fp32) -> bf16 P ----
  for (int r = tid; r < LPAD; r += 256) {
    float mx = -3.4e38f;
    for (int s2 = 0; s2 < SPAD; ++s2) mx = fmaxf(mx, sP[r * SPAD + s2]);
    float sum = 0.0f;
    for (int s2 = 0; s2 < SPAD; ++s2) {
      float e = __expf(sP[r * SPAD + s2] - mx);
      sP[r * SPAD + s2] = e; sum += e;
    }
    float inv = 1.0f / sum;
    for (int s2 = 0; s2 < SPAD; ++s2)
      sPb[r * SPAD + s2] = (__bf16)(sP[r * SPAD + s2] * inv);
  }
  __syncthreads();

  // ---- phase 4: out = P @ V ----
  for (int t = wave; t < LT * 2; t += 8) {
    const int m0 = (t >> 1) * 16, n0 = (t & 1) * 16;
    const int m = m0 + mrow, e = n0 + mrow;
    v8f acc = {};
    for (int kc = 0; kc < SPAD / 32; ++kc) {
      v16bf a = lds_a_frag(sPb + m * SPAD + kc * 32 + khalf);
      v16bf b = lds_b_contig(sVt + e * SPAD + kc * 32 + kb16);  // V^T row: contiguous
      acc = wmma_bf16(a, b, acc);
    }
#pragma unroll
    for (int j = 0; j < 8; ++j) {
      int mm = m0 + j + mhalf;
      if (mm < Lr)
        outp[(long)bq * ob + (long)h * oh + (long)mm * ol + e] = acc[j];
    }
  }
}

// ---------------------------------------------------------------------------
// Y = [LN_{g,b}]( A @ W + bias [+ Res] ), A: Mx256 row-major, Wp packed bf16.
// 32 rows/block; REQUIRES M % 32 == 0. In-place safe.
// ---------------------------------------------------------------------------
__global__ __launch_bounds__(256) void oproj_kernel(
    const float* __restrict__ A, const __bf16* __restrict__ Wp,
    const float* __restrict__ bias, const float* __restrict__ Res,
    const float* __restrict__ lng, const float* __restrict__ lnb,
    float* __restrict__ Y, int M)
{
  __shared__ __align__(16) __bf16 sA[32 * 256];
  __shared__ __align__(16) float  sY[32 * 256];
  const int tid = threadIdx.x, lane = tid & 31, wave = tid >> 5;
  const long r0 = (long)blockIdx.x * 32;
  const int mrow = lane & 15;
  const int khalf = (lane >> 4) << 3, kb16 = (lane >> 4) << 4, mhalf = (lane >> 4) << 3;
  (void)kb16;

  for (int q4 = tid; q4 < 32 * 64; q4 += 256) {     // float4 staging
    int r = q4 >> 6, c = (q4 & 63) << 2;
    f4 v = *(const f4*)(A + (r0 + r) * 256 + c);
#pragma unroll
    for (int i = 0; i < 4; ++i) sA[r * 256 + c + i] = (__bf16)v[i];
  }
  __syncthreads();

  for (int t = wave; t < 32; t += 8) {              // 2 m-tiles x 16 n-tiles
    const int m0 = (t >> 4) * 16, nt = t & 15;
    const int m = m0 + mrow, n = nt * 16 + mrow;
    v8f acc = {};
    for (int kc = 0; kc < 8; ++kc) {
      v16bf a = lds_a_frag(sA + m * 256 + kc * 32 + khalf);
      v16bf b = pk_b_frag(Wp, kc, 16, nt, lane);
      acc = wmma_bf16(a, b, acc);
    }
#pragma unroll
    for (int j = 0; j < 8; ++j)
      sY[(m0 + j + mhalf) * 256 + n] = acc[j];
  }
  __syncthreads();

  // bias + optional residual (coalesced pass)
  for (int idx = tid; idx < 32 * 256; idx += 256) {
    int r = idx >> 8, c = idx & 255;
    float v = sY[idx] + bias[c];
    if (Res != nullptr) v += Res[(r0 + r) * 256 + c];
    sY[idx] = v;
  }
  __syncthreads();

  if (lng != nullptr) {
    for (int r = tid; r < 32; r += 256) {
      float mu = 0.0f;
      for (int c = 0; c < 256; ++c) mu += sY[r * 256 + c];
      mu *= (1.0f / 256.0f);
      float var = 0.0f;
      for (int c = 0; c < 256; ++c) { float d = sY[r * 256 + c] - mu; var += d * d; }
      var *= (1.0f / 256.0f);
      float inv = rsqrtf(var + 1e-5f);
      for (int c = 0; c < 256; ++c)
        sY[r * 256 + c] = (sY[r * 256 + c] - mu) * inv * lng[c] + lnb[c];
    }
    __syncthreads();
  }

  for (int q4 = tid; q4 < 32 * 64; q4 += 256) {     // float4 writeback
    int r = q4 >> 6, c = (q4 & 63) << 2;
    f4 v;
#pragma unroll
    for (int i = 0; i < 4; ++i) v[i] = sY[r * 256 + c + i];
    *(f4*)(Y + (r0 + r) * 256 + c) = v;
  }
}

// ---------------------------------------------------------------------------
// X = LN( X + GELU(X @ W1 + b1) @ W2 + b2 )  in-place; REQUIRES M % 16 == 0.
// ---------------------------------------------------------------------------
__global__ __launch_bounds__(256) void mlp_kernel(
    float* __restrict__ X,
    const __bf16* __restrict__ W1p, const float* __restrict__ b1,
    const __bf16* __restrict__ W2p, const float* __restrict__ b2,
    const float* __restrict__ lng, const float* __restrict__ lnb, int M)
{
  __shared__ __align__(16) __bf16 sX[16 * 256];
  __shared__ __align__(16) __bf16 sH[16 * 1024];
  __shared__ __align__(16) float  sY[16 * 256];
  const int tid = threadIdx.x, lane = tid & 31, wave = tid >> 5;
  const long r0 = (long)blockIdx.x * 16;
  const int mrow = lane & 15;
  const int khalf = (lane >> 4) << 3, mhalf = (lane >> 4) << 3;

  for (int q4 = tid; q4 < 16 * 64; q4 += 256) {
    int r = q4 >> 6, c = (q4 & 63) << 2;
    f4 v = *(const f4*)(X + (r0 + r) * 256 + c);
#pragma unroll
    for (int i = 0; i < 4; ++i) sX[r * 256 + c + i] = (__bf16)v[i];
  }
  __syncthreads();

  // hidden: 16 x 1024, K = 256
  for (int t = wave; t < 64; t += 8) {
    const int n = t * 16 + mrow;
    v8f acc = {};
    for (int kc = 0; kc < 8; ++kc) {
      v16bf a = lds_a_frag(sX + mrow * 256 + kc * 32 + khalf);
      v16bf b = pk_b_frag(W1p, kc, 64, t, lane);
      acc = wmma_bf16(a, b, acc);
    }
    float bb = b1[n];
#pragma unroll
    for (int j = 0; j < 8; ++j)
      sH[(j + mhalf) * 1024 + n] = (__bf16)gelu_exact(acc[j] + bb);
  }
  __syncthreads();

  // out: 16 x 256, K = 1024
  for (int t = wave; t < 16; t += 8) {
    const int n = t * 16 + mrow;
    v8f acc = {};
    for (int kc = 0; kc < 32; ++kc) {
      v16bf a = lds_a_frag(sH + mrow * 1024 + kc * 32 + khalf);
      v16bf b = pk_b_frag(W2p, kc, 16, t, lane);
      acc = wmma_bf16(a, b, acc);
    }
#pragma unroll
    for (int j = 0; j < 8; ++j)
      sY[(j + mhalf) * 256 + n] = acc[j];
  }
  __syncthreads();

  // bias + residual (global X not yet overwritten), coalesced
  for (int idx = tid; idx < 16 * 256; idx += 256) {
    int r = idx >> 8, c = idx & 255;
    sY[idx] += b2[c] + X[(r0 + r) * 256 + c];
  }
  __syncthreads();

  for (int r = tid; r < 16; r += 256) {
    float mu = 0.0f;
    for (int c = 0; c < 256; ++c) mu += sY[r * 256 + c];
    mu *= (1.0f / 256.0f);
    float var = 0.0f;
    for (int c = 0; c < 256; ++c) { float d = sY[r * 256 + c] - mu; var += d * d; }
    var *= (1.0f / 256.0f);
    float inv = rsqrtf(var + 1e-5f);
    for (int c = 0; c < 256; ++c)
      sY[r * 256 + c] = (sY[r * 256 + c] - mu) * inv * lng[c] + lnb[c];
  }
  __syncthreads();

  for (int q4 = tid; q4 < 16 * 64; q4 += 256) {
    int r = q4 >> 6, c = (q4 & 63) << 2;
    f4 v;
#pragma unroll
    for (int i = 0; i < 4; ++i) v[i] = sY[r * 256 + c + i];
    *(f4*)(X + (r0 + r) * 256 + c) = v;
  }
}

// (B, D1, D2, 256) -> (B, D2, D1, 256); one 256-float row per block.
__global__ __launch_bounds__(256) void transpose_kernel(
    const float* __restrict__ in, float* __restrict__ out, int D1, int D2)
{
  long blk = blockIdx.x;               // = (bi*D1 + d1)*D2 + d2
  int c = threadIdx.x;
  long d2 = blk % D2;
  long rest = blk / D2;
  long d1 = rest % D1;
  long bi = rest / D1;
  out[(((bi * D2 + d2) * D1) + d1) * 256 + c] = in[blk * 256 + c];
}

extern "C" void kernel_launch(void* const* d_in, const int* in_sizes, int n_in,
                              void* d_out, int out_size, void* d_ws, size_t ws_size,
                              hipStream_t stream) {
  (void)in_sizes; (void)n_in; (void)out_size; (void)ws_size;
  const float* x    = (const float*)d_in[0];
  const float* tqW  = (const float*)d_in[1];  const float* tqb = (const float*)d_in[2];
  const float* tkW  = (const float*)d_in[3];  const float* tkb = (const float*)d_in[4];
  const float* tvW  = (const float*)d_in[5];  const float* tvb = (const float*)d_in[6];
  const float* toW  = (const float*)d_in[7];  const float* tob = (const float*)d_in[8];
  const float* sqW  = (const float*)d_in[9];  const float* sqb = (const float*)d_in[10];
  const float* skW  = (const float*)d_in[11]; const float* skb = (const float*)d_in[12];
  const float* svW  = (const float*)d_in[13]; const float* svb = (const float*)d_in[14];
  const float* soW  = (const float*)d_in[15]; const float* sob = (const float*)d_in[16];
  const float* rqW  = (const float*)d_in[17]; const float* rqb = (const float*)d_in[18];
  const float* rkW  = (const float*)d_in[19]; const float* rkb = (const float*)d_in[20];
  const float* rvW  = (const float*)d_in[21]; const float* rvb = (const float*)d_in[22];
  const float* roW  = (const float*)d_in[23]; const float* rob = (const float*)d_in[24];
  const float* router = (const float*)d_in[25];
  const float* ln1g = (const float*)d_in[26]; const float* ln1b = (const float*)d_in[27];
  const float* ln2g = (const float*)d_in[28]; const float* ln2b = (const float*)d_in[29];
  const float* ln3g = (const float*)d_in[30]; const float* ln3b = (const float*)d_in[31];
  const float* ln4g = (const float*)d_in[32]; const float* ln4b = (const float*)d_in[33];
  const float* m1W1 = (const float*)d_in[34]; const float* m1b1 = (const float*)d_in[35];
  const float* m1W2 = (const float*)d_in[36]; const float* m1b2 = (const float*)d_in[37];
  const float* m2W1 = (const float*)d_in[38]; const float* m2b1 = (const float*)d_in[39];
  const float* m2W2 = (const float*)d_in[40]; const float* m2b2 = (const float*)d_in[41];
  float* out = (float*)d_out;

  char* ws = (char*)d_ws;
  float* bufA = (float*)(ws);                        // 131072*256 f32 = 134 MB
  float* bufB = (float*)(ws + 134217728ull);         // 134 MB
  float* bufC = (float*)(ws + 268435456ull);         // 10.5 MB
  float* bufD = (float*)(ws + 278921216ull);         // 10.5 MB
  __bf16* wp  = (__bf16*)(ws + 289406976ull);        // packed weights, bf16
  // 12 x (256x256) then 4 MLP matrices
  __bf16* tqWp = wp;                 __bf16* tkWp = wp + 65536;
  __bf16* tvWp = wp + 131072;        __bf16* toWp = wp + 196608;
  __bf16* sqWp = wp + 262144;        __bf16* skWp = wp + 327680;
  __bf16* svWp = wp + 393216;        __bf16* soWp = wp + 458752;
  __bf16* rqWp = wp + 524288;        __bf16* rkWp = wp + 589824;
  __bf16* rvWp = wp + 655360;        __bf16* roWp = wp + 720896;
  __bf16* m1W1p = wp + 786432;                       // 256x1024
  __bf16* m1W2p = m1W1p + 262144;                    // 1024x256
  __bf16* m2W1p = m1W2p + 262144;
  __bf16* m2W2p = m2W1p + 262144;

  // ---- pack all weights to bf16 fragment layout ----
  pack_w_kernel<<<256, 256, 0, stream>>>(tqW, tqWp, 256);
  pack_w_kernel<<<256, 256, 0, stream>>>(tkW, tkWp, 256);
  pack_w_kernel<<<256, 256, 0, stream>>>(tvW, tvWp, 256);
  pack_w_kernel<<<256, 256, 0, stream>>>(toW, toWp, 256);
  pack_w_kernel<<<256, 256, 0, stream>>>(sqW, sqWp, 256);
  pack_w_kernel<<<256, 256, 0, stream>>>(skW, skWp, 256);
  pack_w_kernel<<<256, 256, 0, stream>>>(svW, svWp, 256);
  pack_w_kernel<<<256, 256, 0, stream>>>(soW, soWp, 256);
  pack_w_kernel<<<256, 256, 0, stream>>>(rqW, rqWp, 256);
  pack_w_kernel<<<256, 256, 0, stream>>>(rkW, rkWp, 256);
  pack_w_kernel<<<256, 256, 0, stream>>>(rvW, rvWp, 256);
  pack_w_kernel<<<256, 256, 0, stream>>>(roW, roWp, 256);
  pack_w_kernel<<<1024, 256, 0, stream>>>(m1W1, m1W1p, 1024);
  pack_w_kernel<<<1024, 256, 0, stream>>>(m1W2, m1W2p, 256);
  pack_w_kernel<<<1024, 256, 0, stream>>>(m2W1, m2W1p, 1024);
  pack_w_kernel<<<1024, 256, 0, stream>>>(m2W2, m2W2p, 256);

  const float scale = 0.1767766952966369f;           // 1/sqrt(32)
  const int M1 = 131072;                             // B*TS_D*SEG

  // Stage 1: time attention over SEG: 2048 batches, L=S=64
  dim3 g1(2048, 8);
  attn_kernel<64, 64><<<g1, 256, 0, stream>>>(
      x, 2048, 64 * 256, 64, x, 64 * 256, 64,
      tqWp, tqb, tkWp, tkb, tvWp, tvb,
      bufA, 64 * 256, 32, 256, scale);               // l-major out (mix=False)

  oproj_kernel<<<M1 / 32, 256, 0, stream>>>(bufA, toWp, tob, x, ln1g, ln1b, bufA, M1);
  mlp_kernel<<<M1 / 16, 256, 0, stream>>>(bufA, m1W1p, m1b1, m1W2p, m1b2, ln2g, ln2b, M1);

  // dim_send: (b, TS_D, SEG, DM) -> (b, SEG, TS_D, DM)
  transpose_kernel<<<M1, 256, 0, stream>>>(bufA, bufB, 128, 64);

  // Stage 2a: sender — Q=router (L=10 pad16), KV=dim_send (S=128); 1024 batches
  dim3 g2(1024, 8);
  attn_kernel<16, 128><<<g2, 256, 0, stream>>>(
      router, 64, 10 * 256, 10, bufB, 128 * 256, 128,
      sqWp, sqb, skWp, skb, svWp, svb,
      bufC, 2560, 320, 32, scale);                   // h-major out (mix=True)
  oproj_kernel<<<10240 / 32, 256, 0, stream>>>(bufC, soWp, sob, nullptr, nullptr, nullptr, bufD, 10240);

  // Stage 2b: receiver — Q=dim_send (L=128), KV=dim_buffer (S=10 pad32)
  attn_kernel<128, 32><<<g2, 256, 0, stream>>>(
      bufB, 1024, 128 * 256, 128, bufD, 10 * 256, 10,
      rqWp, rqb, rkWp, rkb, rvWp, rvb,
      bufA, 32768, 4096, 32, scale);                 // h-major out (mix=True)
  oproj_kernel<<<M1 / 32, 256, 0, stream>>>(bufA, roWp, rob, bufB, ln3g, ln3b, bufA, M1);
  mlp_kernel<<<M1 / 16, 256, 0, stream>>>(bufA, m2W1p, m2b1, m2W2p, m2b2, ln4g, ln4b, M1);

  // final: (b, SEG, TS_D, DM) -> (b, TS_D, SEG, DM)
  transpose_kernel<<<M1, 256, 0, stream>>>(bufA, out, 64, 128);
}
